// TriangleAttention_9268539425324
// MI455X (gfx1250) — matrix operand order repaired and checked by reading, not compiled, not run on hardware
//
#include <hip/hip_runtime.h>
#include <hip/hip_fp16.h>

typedef __attribute__((ext_vector_type(16))) _Float16 v16h;
typedef __attribute__((ext_vector_type(8)))  _Float16 v8h;
typedef __attribute__((ext_vector_type(8)))  float    v8f;

#define DEV static __device__ __forceinline__

// ---- WMMA wrapper: D = A(16x32 f16) * B(32x16 f16) + C(16x16 f32) ----
DEV v8f wmma_f16(v16h a, v16h b, v8f c) {
    return __builtin_amdgcn_wmma_f32_16x16x32_f16(
        /*neg_a=*/false, a, /*neg_b=*/false, b,
        /*c_mod=*/(short)0, c, /*reuse_a=*/false, /*reuse_b=*/false);
}

// A-fragment (16 rows x 32 K), source row-major [row][K], stride in halves.
// Lane l<16: row=l, elems 0..7 = K0..7, elems 8..15 = K16..23
// Lane l>=16: row=l-16, elems 0..7 = K8..15, elems 8..15 = K24..31
DEV v16h load_fragA(const _Float16* __restrict__ base, int row0, int stride, int k0) {
    int lane = threadIdx.x & 31;
    int r  = row0 + (lane & 15);
    int hi = lane >> 4;
    const _Float16* p = base + (size_t)r * stride + k0 + hi * 8;
    v8h lo = *(const v8h*)p;
    v8h hb = *(const v8h*)(p + 16);
    v16h f;
#pragma unroll
    for (int t = 0; t < 8; ++t) { f[t] = lo[t]; f[t + 8] = hb[t]; }
    return f;
}

// B-fragment (32 K x 16 cols), source is B^T row-major: [col][K], stride in halves.
// Lane l<16: col=l, elems 0..7 = K0..7, elems 8..15 = K16..23 ; lane l>=16: col=l-16, elems 0..7 = K8..15, elems 8..15 = K24..31
DEV v16h load_fragB(const _Float16* __restrict__ base, int col0, int stride, int k0) {
    int lane = threadIdx.x & 31;
    int n  = col0 + (lane & 15);
    int hi = lane >> 4;
    const _Float16* p = base + (size_t)n * stride + k0 + hi * 8;
    v8h lo = *(const v8h*)p;
    v8h hb = *(const v8h*)(p + 16);
    v16h f;
#pragma unroll
    for (int t = 0; t < 8; ++t) { f[t] = lo[t]; f[t + 8] = hb[t]; }
    return f;
}

// ---------------- LayerNorm: one wave per position, C=128 ----------------
__global__ void ln_kernel(const float* __restrict__ x, const float* __restrict__ w,
                          const float* __restrict__ b, _Float16* __restrict__ xn) {
    int row  = blockIdx.x;
    int lane = threadIdx.x;
    const float* xr = x + (size_t)row * 128;
    float v0[4], s = 0.f, s2 = 0.f;
#pragma unroll
    for (int t = 0; t < 4; ++t) {
        float v = xr[lane + 32 * t];
        v0[t] = v; s += v; s2 += v * v;
    }
#pragma unroll
    for (int off = 16; off; off >>= 1) {
        s  += __shfl_xor(s,  off, 32);
        s2 += __shfl_xor(s2, off, 32);
    }
    float mu  = s * (1.f / 128.f);
    float var = s2 * (1.f / 128.f) - mu * mu;
    float rs  = rsqrtf(var + 1e-5f);
#pragma unroll
    for (int t = 0; t < 4; ++t) {
        int c = lane + 32 * t;
        xn[(size_t)row * 128 + c] = (_Float16)((v0[t] - mu) * rs * w[c] + b[c]);
    }
}

// ---------------- f32 -> f16 weight conversion ----------------
__global__ void f2h_kernel(const float* __restrict__ src, _Float16* __restrict__ dst, int n) {
    int i = blockIdx.x * 256 + threadIdx.x;
    if (i < n) dst[i] = (_Float16)src[i];
}

// ------------- Projection GEMM: out[65536,128] = xn * W^T ---------------
// One wave computes a full 16x128 row-strip: A-fragments hoisted, 32 WMMAs.
// mode 0: Q (scale, row-major f16) | 1: K (row-major f16)
// mode 2: V transposed -> Vt[(m*4+h)*32 + d][j]   | 3: G (sigmoid(x+bg), f16)
__global__ void proj_kernel(const _Float16* __restrict__ xn, const _Float16* __restrict__ wh,
                            const float* __restrict__ bg, _Float16* __restrict__ dst,
                            int mode, float scale) {
    int mt = blockIdx.x;
    int lane = threadIdx.x;
    v16h A[4];
#pragma unroll
    for (int kc = 0; kc < 4; ++kc) A[kc] = load_fragA(xn, mt * 16, 128, kc * 32);

    v8f acc[8];
#pragma unroll
    for (int nt = 0; nt < 8; ++nt) {
        v8f c = {};
#pragma unroll
        for (int kc = 0; kc < 4; ++kc) {
            v16h b = load_fragB(wh, nt * 16, 128, kc * 32);
            c = wmma_f16(A[kc], b, c);
        }
        acc[nt] = c;
    }

    int hi = lane >> 4, cl = lane & 15;
#pragma unroll
    for (int nt = 0; nt < 8; ++nt) {
        int col = nt * 16 + cl;
#pragma unroll
        for (int r = 0; r < 8; ++r) {
            int row = mt * 16 + r + 8 * hi;
            float v = acc[nt][r] * scale;
            if (mode == 3) {
                v += bg[col];
                v = __builtin_amdgcn_rcpf(1.f + __expf(-v));   // fast sigmoid
            }
            _Float16 hv = (_Float16)v;
            if (mode == 2) {
                int m = row >> 8, j = row & 255, h = col >> 5, d = col & 31;
                dst[(((size_t)(m * 4 + h)) * 32 + d) * 256 + j] = hv;
            } else {
                dst[(size_t)row * 128 + col] = hv;
            }
        }
    }
}

// ------------- Bias projection: biasb[h][i][j] = xn[i*256+j] . wb[h] -------------
__global__ void biasproj_kernel(const _Float16* __restrict__ xn, const float* __restrict__ wb,
                                float* __restrict__ biasb) {
    int pos = blockIdx.x;
    int lane = threadIdx.x;
    float acc[4] = {0.f, 0.f, 0.f, 0.f};
#pragma unroll
    for (int t = 0; t < 4; ++t) {
        int c = lane + 32 * t;
        float xv = (float)xn[(size_t)pos * 128 + c];
#pragma unroll
        for (int h = 0; h < 4; ++h) acc[h] += xv * wb[h * 128 + c];
    }
#pragma unroll
    for (int h = 0; h < 4; ++h) {
#pragma unroll
        for (int off = 16; off; off >>= 1) acc[h] += __shfl_xor(acc[h], off, 32);
    }
    if (lane == 0) {
#pragma unroll
        for (int h = 0; h < 4; ++h) biasb[(size_t)h * 65536 + pos] = acc[h];
    }
}

// ------------- Attention: one wave per (m, h, 16-row i-tile) -------------
__global__ void attn_kernel(const _Float16* __restrict__ Qb, const _Float16* __restrict__ Kb,
                            const _Float16* __restrict__ Vt, const _Float16* __restrict__ Gb,
                            const float* __restrict__ biasb, _Float16* __restrict__ Ob) {
    __shared__ __align__(16) float    bias_s[16 * 256];
    __shared__ __align__(16) _Float16 w_s[16 * 256];
    int bid = blockIdx.x;
    int m  = bid >> 6;
    int h  = (bid >> 4) & 3;
    int it = bid & 15;
    int lane = threadIdx.x;
    int hi = lane >> 4, cl = lane & 15;

    // bias tile [16 x 256] contiguous in biasb -> async copy global->LDS (no VGPR staging)
    {
        unsigned lds_base = (unsigned)(size_t)(void*)&bias_s[0];
        const char* src = (const char*)(biasb + (size_t)h * 65536 + (size_t)it * 4096);
#pragma unroll
        for (int t = 0; t < 32; ++t) {
            unsigned    ldsa = lds_base + (unsigned)((lane + 32 * t) * 16);
            const char* ga   = src + (size_t)(lane + 32 * t) * 16;
            asm volatile("global_load_async_to_lds_b128 %0, %1, off"
                         :: "v"(ldsa), "v"(ga) : "memory");
        }
        asm volatile("s_wait_asynccnt 0x0" ::: "memory");
    }

    // S = (Q/sqrt(D)) K^T + bias : 16 WMMAs over j-tiles, K dim = head dim = 32
    v16h aq = load_fragA(Qb, m * 256 + it * 16, 128, h * 32);
    v8f S[16];
#pragma unroll
    for (int jt = 0; jt < 16; ++jt) {
        v16h bk = load_fragB(Kb, m * 256 + jt * 16, 128, h * 32);
        v8f z = {};
        S[jt] = wmma_f16(aq, bk, z);
#pragma unroll
        for (int r = 0; r < 8; ++r)
            S[jt][r] += bias_s[(r + 8 * hi) * 256 + jt * 16 + cl];
    }

    // Row softmax. Each row lives in one 16-lane half (hi selects rows 0-7 / 8-15).
#pragma unroll
    for (int r = 0; r < 8; ++r) {
        float mx = -3.0e38f;
#pragma unroll
        for (int jt = 0; jt < 16; ++jt) mx = fmaxf(mx, S[jt][r]);
#pragma unroll
        for (int off = 1; off < 16; off <<= 1) mx = fmaxf(mx, __shfl_xor(mx, off, 32));
        float sum = 0.f;
#pragma unroll
        for (int jt = 0; jt < 16; ++jt) {
            float e = __expf(S[jt][r] - mx);
            S[jt][r] = e; sum += e;
        }
#pragma unroll
        for (int off = 1; off < 16; off <<= 1) sum += __shfl_xor(sum, off, 32);
        float inv = __builtin_amdgcn_rcpf(sum);
#pragma unroll
        for (int jt = 0; jt < 16; ++jt) S[jt][r] *= inv;
    }

    // C-layout -> LDS (row-major weights) so we can refetch in A-layout.
#pragma unroll
    for (int jt = 0; jt < 16; ++jt)
#pragma unroll
        for (int r = 0; r < 8; ++r)
            w_s[(r + 8 * hi) * 256 + jt * 16 + cl] = (_Float16)S[jt][r];

    // O[16 x 32] = W[16 x 256] * V[256 x 32]; Vt is B^T so fragB loads are contiguous.
    const _Float16* vt = Vt + ((size_t)(m * 4 + h)) * 32 * 256;
    v8f O0 = {}, O1 = {};
#pragma unroll
    for (int kc = 0; kc < 8; ++kc) {
        const _Float16* p = &w_s[cl * 256 + kc * 32 + hi * 8];
        v8h lo = *(const v8h*)p;
        v8h hb = *(const v8h*)(p + 16);
        v16h aw;
#pragma unroll
        for (int t = 0; t < 8; ++t) { aw[t] = lo[t]; aw[t + 8] = hb[t]; }
        v16h bv0 = load_fragB(vt, 0,  256, kc * 32);
        v16h bv1 = load_fragB(vt, 16, 256, kc * 32);
        O0 = wmma_f16(aw, bv0, O0);
        O1 = wmma_f16(aw, bv1, O1);
    }

    // gate with sigmoid(g) and store f16
#pragma unroll
    for (int nt = 0; nt < 2; ++nt) {
        v8f& O = nt ? O1 : O0;
#pragma unroll
        for (int r = 0; r < 8; ++r) {
            int row = m * 256 + it * 16 + r + 8 * hi;
            int col = h * 32 + nt * 16 + cl;
            float g = (float)Gb[(size_t)row * 128 + col];
            Ob[(size_t)row * 128 + col] = (_Float16)(O[r] * g);
        }
    }
}

// ------------- Output projection: out = Ob * wo^T + bo (f32 out) -------------
// One wave computes a 16x128 row-strip (32 WMMAs), A-fragments hoisted.
__global__ void outproj_kernel(const _Float16* __restrict__ Ob, const _Float16* __restrict__ woh,
                               const float* __restrict__ bo, float* __restrict__ out) {
    int mt = blockIdx.x;
    int lane = threadIdx.x;
    int hi = lane >> 4, cl = lane & 15;

    v16h A[4];
#pragma unroll
    for (int kc = 0; kc < 4; ++kc) A[kc] = load_fragA(Ob, mt * 16, 128, kc * 32);

#pragma unroll
    for (int nt = 0; nt < 8; ++nt) {
        v8f acc = {};
#pragma unroll
        for (int kc = 0; kc < 4; ++kc) {
            v16h b = load_fragB(woh, nt * 16, 128, kc * 32);
            acc = wmma_f16(A[kc], b, acc);
        }
        int col = nt * 16 + cl;
        float bb = bo[col];
#pragma unroll
        for (int r = 0; r < 8; ++r)
            out[(size_t)(mt * 16 + r + 8 * hi) * 128 + col] = acc[r] + bb;
    }
}

extern "C" void kernel_launch(void* const* d_in, const int* in_sizes, int n_in,
                              void* d_out, int out_size, void* d_ws, size_t ws_size,
                              hipStream_t stream) {
    (void)in_sizes; (void)n_in; (void)out_size; (void)ws_size;
    const float* x2d = (const float*)d_in[0];
    const float* nw  = (const float*)d_in[1];
    const float* nb  = (const float*)d_in[2];
    const float* wq  = (const float*)d_in[3];
    const float* wk  = (const float*)d_in[4];
    const float* wv  = (const float*)d_in[5];
    const float* wb  = (const float*)d_in[6];
    const float* wg  = (const float*)d_in[7];
    const float* bg  = (const float*)d_in[8];
    const float* wo  = (const float*)d_in[9];
    const float* bo  = (const float*)d_in[10];
    float* out = (float*)d_out;

    const size_t P = 65536;            // N*N positions
    char* ws = (char*)d_ws;
    _Float16* xn  = (_Float16*)ws; ws += P * 128 * 2;   // 16 MB
    _Float16* Qb  = (_Float16*)ws; ws += P * 128 * 2;
    _Float16* Kb  = (_Float16*)ws; ws += P * 128 * 2;
    _Float16* Vt  = (_Float16*)ws; ws += P * 128 * 2;
    _Float16* Gb  = (_Float16*)ws; ws += P * 128 * 2;
    _Float16* Ob  = (_Float16*)ws; ws += P * 128 * 2;
    float*    biasb = (float*)ws;  ws += 4 * P * 4;     // 1 MB
    _Float16* wqh = (_Float16*)ws; ws += 16384 * 2;
    _Float16* wkh = (_Float16*)ws; ws += 16384 * 2;
    _Float16* wvh = (_Float16*)ws; ws += 16384 * 2;
    _Float16* wgh = (_Float16*)ws; ws += 16384 * 2;
    _Float16* woh = (_Float16*)ws; ws += 16384 * 2;

    dim3 b32(32);
    ln_kernel<<<65536, b32, 0, stream>>>(x2d, nw, nb, xn);

    f2h_kernel<<<64, 256, 0, stream>>>(wq, wqh, 16384);
    f2h_kernel<<<64, 256, 0, stream>>>(wk, wkh, 16384);
    f2h_kernel<<<64, 256, 0, stream>>>(wv, wvh, 16384);
    f2h_kernel<<<64, 256, 0, stream>>>(wg, wgh, 16384);
    f2h_kernel<<<64, 256, 0, stream>>>(wo, woh, 16384);

    const float qscale = 0.17677669529663687f;  // 1/sqrt(32)
    proj_kernel<<<4096, b32, 0, stream>>>(xn, wqh, nullptr, Qb, 0, qscale);
    proj_kernel<<<4096, b32, 0, stream>>>(xn, wkh, nullptr, Kb, 1, 1.0f);
    proj_kernel<<<4096, b32, 0, stream>>>(xn, wvh, nullptr, Vt, 2, 1.0f);
    proj_kernel<<<4096, b32, 0, stream>>>(xn, wgh, bg,      Gb, 3, 1.0f);

    biasproj_kernel<<<65536, b32, 0, stream>>>(xn, wb, biasb);

    attn_kernel<<<16384, b32, 0, stream>>>(Qb, Kb, Vt, Gb, biasb, Ob);

    outproj_kernel<<<4096, b32, 0, stream>>>(Ob, woh, bo, out);
}
